// BertLayerWithMoE_27848567947630
// MI455X (gfx1250) — matrix operand order repaired and checked by reading, not compiled, not run on hardware
//
#include <hip/hip_runtime.h>
#include <hip/hip_bf16.h>
#include <math.h>

#define H_  1024
#define NH_ 16
#define DH_ 64
#define I_  4096
#define E_  8
#define B_  2
#define S_  2048
#define N_  (B_*S_)

typedef __attribute__((ext_vector_type(16))) __bf16       v16bf;
typedef __attribute__((ext_vector_type(8)))  float        v8f;
typedef __attribute__((ext_vector_type(4)))  float        vf4;
typedef __attribute__((ext_vector_type(4)))  unsigned int v4u;
typedef __attribute__((ext_vector_type(8)))  int          v8i;
typedef __attribute__((ext_vector_type(4)))  int          v4i;

union BFrag { v16bf v; __bf16 h[16]; unsigned int u[8]; };
union FAcc  { v8f v; float f[8]; };
union Pack8 { v4u u4; __bf16 h[8]; };
union Pack4 { vf4 v; float f[4]; };

__device__ __forceinline__ __bf16 f2bf(float x) { return (__bf16)x; }
__device__ __forceinline__ float  bf2f(__bf16 x) { return (float)x; }
// Raw v_exp_f32 (no libm subnormal fixup); args here are always max-subtracted.
__device__ __forceinline__ float fast_exp2(float x) { return __builtin_amdgcn_exp2f(x); }

__device__ __forceinline__ v8f wmma_bf16(v16bf a, v16bf b, v8f c) {
  return __builtin_amdgcn_wmma_f32_16x16x32_bf16(false, a, false, b, (short)0, c,
                                                 false, false);
}

// A-operand (16-bit, 16x32): VGPR pair j -> K index (lanes 0-15: K0-15, 16-31: K16-31)
__device__ __forceinline__ int akp(int j, int lh) {
  return j * 2 + lh * 16;
}
// B-operand (16-bit, 32x16): VGPR pair j -> K index (lanes 0-15: K0-15, 16-31: K16-31)
__device__ __forceinline__ int bkp(int j, int lh) {
  return j * 2 + lh * 16;
}

// ---------------------------------------------------------------------------
// Tensor Data Mover: 2D tile (tile0 x tile1) from a row-major tensor into LDS.
// dsize_code: 0=1B 1=2B 2=4B. Dims/strides in element units (ISA 8.3/8.4).
// ---------------------------------------------------------------------------
__device__ __forceinline__ void tdm_load_2d(unsigned int lds_addr,
                                            unsigned long long ga,
                                            int dsize_code,
                                            unsigned int tdim0, unsigned int tdim1,
                                            unsigned int tile0, unsigned int tile1,
                                            unsigned int stride0) {
  v4u g0;
  g0[0] = 1u;                                          // count=1, user descriptor
  g0[1] = lds_addr;
  g0[2] = (unsigned int)(ga & 0xffffffffull);
  g0[3] = (unsigned int)((ga >> 32) & 0x01ffffffull) | 0x80000000u;  // type=2
  v8i g1;
  g1[0] = dsize_code << 16;                            // data_size, no multicast
  g1[1] = (int)((tdim0 & 0xffffu) << 16);
  g1[2] = (int)(((tdim0 >> 16) & 0xffffu) | ((tdim1 & 0xffffu) << 16));
  g1[3] = (int)(((tdim1 >> 16) & 0xffffu) | ((tile0 & 0xffffu) << 16));
  g1[4] = (int)(tile1 & 0xffffu);                      // tile_dim1 (tile_dim2=0)
  g1[5] = (int)stride0;                                // tensor_dim0_stride
  g1[6] = 0; g1[7] = 0;
  v4i gz = {0, 0, 0, 0};
#if __clang_major__ >= 23
  v8i gz8 = {0, 0, 0, 0, 0, 0, 0, 0};
  __builtin_amdgcn_tensor_load_to_lds(g0, g1, gz, gz, gz8, 0);
#else
  __builtin_amdgcn_tensor_load_to_lds(g0, g1, gz, gz, 0);
#endif
}

// ---------------------------------------------------------------------------
// Generic tiled GEMM: C = (A @ W + bias [+ res]) * oscale, bf16 WMMA core.
// Tile 64x64, BK=64 (two WMMA k-steps per barrier round).
// ---------------------------------------------------------------------------
template<bool ABF16, bool OUTBF16, bool ADDRES>
__global__ __launch_bounds__(256)
void k_gemm(const void* __restrict__ Ap, const float* __restrict__ W,
            const float* __restrict__ bias, const float* __restrict__ res,
            void* __restrict__ Cp, int M, int Nn, int K, float oscale)
{
  __shared__ __align__(16) __bf16 sA[64 * 64];
  __shared__ __align__(16) __bf16 sB[64 * 64];   // [n][k]
  const int tid = threadIdx.x, lane = tid & 31, wave = tid >> 5;
  const int lh = lane >> 4, l16 = lane & 15;
  const int n0 = blockIdx.x * 64, m0 = blockIdx.y * 64;
  const int wm = wave & 3, wn = wave >> 2;
  const float*  Af = (const float*)Ap;
  const __bf16* Ab = (const __bf16*)Ap;

  FAcc acc0, acc1;
  #pragma unroll
  for (int j = 0; j < 8; ++j) { acc0.f[j] = 0.f; acc1.f[j] = 0.f; }

  for (int k0 = 0; k0 < K; k0 += 64) {
    #pragma unroll
    for (int it = 0; it < 2; ++it) {           // A tile 64x64 -> bf16 (b128)
      int ch = tid + it * 256;
      int r = ch >> 3, c = (ch & 7) * 8;
      if (ABF16) {
        *(v4u*)&sA[r * 64 + c] = *(const v4u*)&Ab[(size_t)(m0 + r) * K + k0 + c];
      } else {
        Pack4 a0, a1; Pack8 p;
        a0.v = *(const vf4*)&Af[(size_t)(m0 + r) * K + k0 + c];
        a1.v = *(const vf4*)&Af[(size_t)(m0 + r) * K + k0 + c + 4];
        #pragma unroll
        for (int i = 0; i < 4; ++i) { p.h[i] = f2bf(a0.f[i]); p.h[i+4] = f2bf(a1.f[i]); }
        *(v4u*)&sA[r * 64 + c] = p.u4;
      }
    }
    #pragma unroll
    for (int it = 0; it < 2; ++it) {           // W tile 64x64 -> bf16, [n][k]
      int ch = tid + it * 256;
      int r = ch >> 3, c = (ch & 7) * 8;       // r = k row, c = n col base
      Pack4 w0, w1;
      w0.v = *(const vf4*)&W[(size_t)(k0 + r) * Nn + n0 + c];
      w1.v = *(const vf4*)&W[(size_t)(k0 + r) * Nn + n0 + c + 4];
      #pragma unroll
      for (int i = 0; i < 4; ++i) {
        sB[(c + i) * 64 + r]     = f2bf(w0.f[i]);
        sB[(c + 4 + i) * 64 + r] = f2bf(w1.f[i]);
      }
    }
    if (k0 + 64 < K)
      __builtin_prefetch(&W[(size_t)(k0 + 64) * Nn + n0 + (tid & 63)], 0, 1);
    __syncthreads();

    #pragma unroll
    for (int ks = 0; ks < 2; ++ks) {
      BFrag af;
      #pragma unroll
      for (int j = 0; j < 8; ++j)
        af.u[j] = *(const unsigned int*)&sA[(wm * 16 + l16) * 64 + ks * 32 + akp(j, lh)];
      BFrag bf0, bf1;
      #pragma unroll
      for (int j = 0; j < 8; ++j) {
        int kk = ks * 32 + bkp(j, lh);
        bf0.u[j] = *(const unsigned int*)&sB[(wn * 32 + l16) * 64 + kk];
        bf1.u[j] = *(const unsigned int*)&sB[(wn * 32 + 16 + l16) * 64 + kk];
      }
      acc0.v = wmma_bf16(af.v, bf0.v, acc0.v);
      acc1.v = wmma_bf16(af.v, bf1.v, acc1.v);
    }
    __syncthreads();
  }

  #pragma unroll
  for (int j = 0; j < 8; ++j) {
    int m  = m0 + wm * 16 + j + lh * 8;
    int nA = n0 + wn * 32 + l16;
    int nB = nA + 16;
    float vA = acc0.f[j] + bias[nA];
    float vB = acc1.f[j] + bias[nB];
    if constexpr (ADDRES) {
      vA += res[(size_t)m * Nn + nA];
      vB += res[(size_t)m * Nn + nB];
    }
    vA *= oscale; vB *= oscale;
    if constexpr (OUTBF16) {
      ((__bf16*)Cp)[(size_t)m * Nn + nA] = f2bf(vA);
      ((__bf16*)Cp)[(size_t)m * Nn + nB] = f2bf(vB);
    } else {
      ((float*)Cp)[(size_t)m * Nn + nA] = vA;
      ((float*)Cp)[(size_t)m * Nn + nB] = vB;
    }
  }
}

// ---------------------------------------------------------------------------
// Flash attention. Q is pre-scaled by 0.125*log2(e); softmax runs in exp2
// domain. S^T = K Q^T, O^T = V^T P^T. Q/K tiles staged by the TDM.
// ---------------------------------------------------------------------------
__global__ __launch_bounds__(256)
void k_attn(const __bf16* __restrict__ q, const __bf16* __restrict__ kk_,
            const __bf16* __restrict__ v, float* __restrict__ ctx)
{
  __shared__ __align__(16) __bf16 sQ[128 * 64];
  __shared__ __align__(16) __bf16 sK[64 * 64];
  __shared__ __align__(16) __bf16 sV[64 * 64];   // transposed [dh][key]
  const int tid = threadIdx.x, lane = tid & 31, wave = tid >> 5;
  const int lh = lane >> 4, l16 = lane & 15;
  const int qbase = blockIdx.x * 128;
  const int h = blockIdx.y, b = blockIdx.z;
  const size_t hd = (size_t)h * DH_;

  if (wave == 0) {                               // TDM: Q tile 128x64 bf16
    unsigned long long qa = (unsigned long long)(size_t)(const void*)q +
        ((size_t)(b * S_ + qbase) * H_ + hd) * 2ull;
    tdm_load_2d((unsigned int)(size_t)(void*)&sQ[0], qa, 1, H_, N_, 64, 128, H_);
    __builtin_amdgcn_s_wait_tensorcnt(0);
  }
  __syncthreads();

  BFrag bq0, bq1;                  // Q^T B-fragments, dh k-steps 0..31 / 32..63
  #pragma unroll
  for (int j = 0; j < 8; ++j) {
    int kkp = bkp(j, lh);
    bq0.u[j] = *(const unsigned int*)&sQ[(wave * 16 + l16) * 64 + kkp];
    bq1.u[j] = *(const unsigned int*)&sQ[(wave * 16 + l16) * 64 + 32 + kkp];
  }

  const unsigned long long kbase = (unsigned long long)(size_t)(const void*)kk_ +
      ((size_t)(b * S_) * H_ + hd) * 2ull;

  float mrow = -3.0e38f, lrow = 0.f;
  FAcc O[4];
  #pragma unroll
  for (int t = 0; t < 4; ++t)
    for (int j = 0; j < 8; ++j) O[t].f[j] = 0.f;

  for (int kc = 0; kc < S_ / 64; ++kc) {
    __syncthreads();
    const int key0 = kc * 64;
    if (wave == 0)                               // TDM: K tile 64x64 bf16
      tdm_load_2d((unsigned int)(size_t)(void*)&sK[0],
                  kbase + (unsigned long long)key0 * (H_ * 2), 1, H_, N_, 64, 64, H_);
    #pragma unroll
    for (int it = 0; it < 2; ++it) {             // V: b128 load + LDS transpose
      int ch = tid + it * 256;
      int r = ch >> 3, c = (ch & 7) * 8;
      Pack8 pv;
      pv.u4 = *(const v4u*)&v[(size_t)(b * S_ + key0 + r) * H_ + hd + c];
      #pragma unroll
      for (int i = 0; i < 8; ++i) sV[(c + i) * 64 + r] = pv.h[i];
    }
    if (wave == 0) __builtin_amdgcn_s_wait_tensorcnt(0);
    __syncthreads();

    #pragma unroll
    for (int sub = 0; sub < 2; ++sub) {
      FAcc s0, s1;
      #pragma unroll
      for (int j = 0; j < 8; ++j) { s0.f[j] = 0.f; s1.f[j] = 0.f; }
      BFrag ak;
      #pragma unroll
      for (int ks = 0; ks < 2; ++ks) {
        #pragma unroll
        for (int j = 0; j < 8; ++j)
          ak.u[j] = *(const unsigned int*)
              &sK[(sub * 32 + l16) * 64 + ks * 32 + akp(j, lh)];
        s0.v = wmma_bf16(ak.v, ks ? bq1.v : bq0.v, s0.v);
        #pragma unroll
        for (int j = 0; j < 8; ++j)
          ak.u[j] = *(const unsigned int*)
              &sK[(sub * 32 + 16 + l16) * 64 + ks * 32 + akp(j, lh)];
        s1.v = wmma_bf16(ak.v, ks ? bq1.v : bq0.v, s1.v);
      }
      float tm = -3.0e38f;                       // scores already in exp2 domain
      #pragma unroll
      for (int j = 0; j < 8; ++j)
        tm = fmaxf(tm, fmaxf(s0.f[j], s1.f[j]));
      tm = fmaxf(tm, __shfl_xor(tm, 16, 32));
      float mnew  = fmaxf(mrow, tm);
      float alpha = fast_exp2(mrow - mnew);
      mrow = mnew;
      float p0[8], p1[8], ls = 0.f;
      #pragma unroll
      for (int j = 0; j < 8; ++j) {
        p0[j] = fast_exp2(s0.f[j] - mnew);
        p1[j] = fast_exp2(s1.f[j] - mnew);
        ls += p0[j] + p1[j];
      }
      lrow = lrow * alpha + ls;
      BFrag pt;                                  // P^T B-fragment via half-swap
      #pragma unroll
      for (int j = 0; j < 8; ++j) {
        float o0 = __shfl_xor(p0[j], 16, 32);
        float o1 = __shfl_xor(p1[j], 16, 32);
        if (lh == 0) { pt.h[j] = f2bf(p0[j]); pt.h[j + 8] = f2bf(o0); }
        else         { pt.h[j] = f2bf(o1);    pt.h[j + 8] = f2bf(p1[j]); }
      }
      #pragma unroll
      for (int t = 0; t < 4; ++t) {
        #pragma unroll
        for (int j = 0; j < 8; ++j) O[t].f[j] *= alpha;
        BFrag av;
        #pragma unroll
        for (int j = 0; j < 8; ++j)
          av.u[j] = *(const unsigned int*)
              &sV[(t * 16 + l16) * 64 + sub * 32 + akp(j, lh)];
        O[t].v = wmma_bf16(av.v, pt.v, O[t].v);
      }
    }
  }
  float ltot = lrow + __shfl_xor(lrow, 16, 32);
  float inv = 1.0f / ltot;
  const int qrow = qbase + wave * 16 + l16;
  float* crow = &ctx[(size_t)(b * S_ + qrow) * H_ + hd];
  #pragma unroll
  for (int t = 0; t < 4; ++t) {                  // rows contiguous in dh: b128 stores
    Pack4 lo, hi;
    #pragma unroll
    for (int j = 0; j < 4; ++j) { lo.f[j] = O[t].f[j] * inv; hi.f[j] = O[t].f[j + 4] * inv; }
    *(vf4*)&crow[t * 16 + lh * 8]     = lo.v;
    *(vf4*)&crow[t * 16 + lh * 8 + 4] = hi.v;
  }
}

// ---------------------------------------------------------------------------
// LayerNorm over last dim (H=1024), one block per row, float4 traffic
// ---------------------------------------------------------------------------
__global__ __launch_bounds__(256)
void k_ln(const float* __restrict__ x, const float* __restrict__ g,
          const float* __restrict__ bb, float* __restrict__ out)
{
  __shared__ float rs[256], rq[256];
  const int row = blockIdx.x, tid = threadIdx.x;
  const float* xr = x + (size_t)row * H_;
  Pack4 vv;
  vv.v = *(const vf4*)&xr[tid * 4];
  float s = 0.f, sq = 0.f;
  #pragma unroll
  for (int i = 0; i < 4; ++i) { s += vv.f[i]; sq += vv.f[i] * vv.f[i]; }
  rs[tid] = s; rq[tid] = sq;
  __syncthreads();
  for (int off = 128; off > 0; off >>= 1) {
    if (tid < off) { rs[tid] += rs[tid + off]; rq[tid] += rq[tid + off]; }
    __syncthreads();
  }
  float mean = rs[0] * (1.0f / H_);
  float var  = rq[0] * (1.0f / H_) - mean * mean;
  float rinv = __builtin_amdgcn_rsqf(var + 1e-12f);
  Pack4 gg, bbv, ov;
  gg.v  = *(const vf4*)&g[tid * 4];
  bbv.v = *(const vf4*)&bb[tid * 4];
  #pragma unroll
  for (int i = 0; i < 4; ++i)
    ov.f[i] = (vv.f[i] - mean) * rinv * gg.f[i] + bbv.f[i];
  *(vf4*)&out[(size_t)row * H_ + tid * 4] = ov.v;
}

// ---------------------------------------------------------------------------
// MoE routing
// ---------------------------------------------------------------------------
__global__ void k_zero16(int* p) { if (threadIdx.x < 16) p[threadIdx.x] = 0; }

__global__ __launch_bounds__(256)
void k_router(const float* __restrict__ x, const float* __restrict__ rw,
              int* __restrict__ idx, int* __restrict__ counts)
{
  int t = blockIdx.x * 256 + threadIdx.x;
  const float* xr = x + (size_t)t * H_;
  float acc[E_];
  #pragma unroll
  for (int e = 0; e < E_; ++e) acc[e] = 0.f;
  for (int kk = 0; kk < H_; kk += 4) {
    Pack4 xv; xv.v = *(const vf4*)&xr[kk];
    #pragma unroll
    for (int i = 0; i < 4; ++i) {
      Pack4 r0, r1;
      r0.v = *(const vf4*)&rw[(kk + i) * E_];
      r1.v = *(const vf4*)&rw[(kk + i) * E_ + 4];
      #pragma unroll
      for (int e = 0; e < 4; ++e) {
        acc[e]     += xv.f[i] * r0.f[e];
        acc[e + 4] += xv.f[i] * r1.f[e];
      }
    }
  }
  int best = 0; float bv = acc[0];
  #pragma unroll
  for (int e = 1; e < E_; ++e) if (acc[e] > bv) { bv = acc[e]; best = e; }
  idx[t] = best;
  atomicAdd(&counts[best], 1);
}

__global__ void k_offsets(const int* counts, int* offsets, int* cursor) {
  if (threadIdx.x == 0) {
    int run = 0;
    for (int e = 0; e < E_; ++e) { offsets[e] = run; cursor[e] = run; run += counts[e]; }
  }
}

__global__ __launch_bounds__(256)
void k_scatter(const int* __restrict__ idx, int* __restrict__ cursor,
               int* __restrict__ perm) {
  int t = blockIdx.x * 256 + threadIdx.x;
  int pos = atomicAdd(&cursor[idx[t]], 1);
  perm[pos] = t;
}

// ---------------------------------------------------------------------------
// Expert up-proj GEMM (gathered rows) + GELU; 64x64 f32 weight tiles via TDM
// ---------------------------------------------------------------------------
__global__ __launch_bounds__(256)
void k_expert(const float* __restrict__ x, const float* __restrict__ We,
              const float* __restrict__ be, const int* __restrict__ perm,
              const int* __restrict__ offsets, const int* __restrict__ counts,
              __bf16* __restrict__ inter)
{
  __shared__ __align__(16) __bf16 sA[64 * 64];
  __shared__ __align__(16) float  sW[64 * 64];   // [k][n], filled by TDM (f32)
  __shared__ int sTok[64];
  const int nb = blockIdx.x, mb = blockIdx.y, e = blockIdx.z;
  const int cnt = counts[e];
  if (mb * 64 >= cnt) return;
  const int base = offsets[e];
  const int tid = threadIdx.x, lane = tid & 31, wave = tid >> 5;
  const int lh = lane >> 4, l16 = lane & 15;
  const int wm = wave & 3, wn = wave >> 2;
  if (tid < 64) {
    int rr = mb * 64 + tid;
    sTok[tid] = perm[base + (rr < cnt ? rr : cnt - 1)];
  }
  __syncthreads();

  FAcc acc0, acc1;
  #pragma unroll
  for (int j = 0; j < 8; ++j) { acc0.f[j] = 0.f; acc1.f[j] = 0.f; }

  const unsigned int ldsW = (unsigned int)(size_t)(void*)&sW[0];
  const unsigned long long wbase =
      (unsigned long long)(size_t)(const void*)We +
      ((unsigned long long)e * H_ * I_ + (unsigned long long)nb * 64) * 4ull;

  for (int k0 = 0; k0 < H_; k0 += 64) {
    __syncthreads();
    if (wave == 0)                               // TDM: 64x64 f32 tile of We[e]
      tdm_load_2d(ldsW, wbase + (unsigned long long)k0 * I_ * 4ull,
                  2, I_, H_, 64, 64, I_);
    #pragma unroll
    for (int it = 0; it < 2; ++it) {             // gathered A tile -> bf16, b128 loads
      int ch = tid + it * 256;
      int r = ch >> 3, c = (ch & 7) * 8;
      Pack4 a0, a1; Pack8 p;
      const float* xr = &x[(size_t)sTok[r] * H_ + k0 + c];
      a0.v = *(const vf4*)&xr[0];
      a1.v = *(const vf4*)&xr[4];
      #pragma unroll
      for (int i = 0; i < 4; ++i) { p.h[i] = f2bf(a0.f[i]); p.h[i+4] = f2bf(a1.f[i]); }
      *(v4u*)&sA[r * 64 + c] = p.u4;
    }
    if (wave == 0) __builtin_amdgcn_s_wait_tensorcnt(0);
    __syncthreads();

    #pragma unroll
    for (int ks = 0; ks < 2; ++ks) {
      BFrag af;
      #pragma unroll
      for (int j = 0; j < 8; ++j)
        af.u[j] = *(const unsigned int*)&sA[(wm * 16 + l16) * 64 + ks * 32 + akp(j, lh)];
      BFrag bf0, bf1;
      #pragma unroll
      for (int i = 0; i < 16; ++i) {             // f32 LDS tile -> bf16 frags
        int kk = ks * 32 + i + lh * 16;
        bf0.h[i] = f2bf(sW[kk * 64 + wn * 32 + l16]);
        bf1.h[i] = f2bf(sW[kk * 64 + wn * 32 + 16 + l16]);
      }
      acc0.v = wmma_bf16(af.v, bf0.v, acc0.v);
      acc1.v = wmma_bf16(af.v, bf1.v, acc1.v);
    }
  }
  __syncthreads();
  #pragma unroll
  for (int j = 0; j < 8; ++j) {
    int r = wm * 16 + j + lh * 8;
    if (mb * 64 + r >= cnt) continue;
    int tok = sTok[r];
    int nA = nb * 64 + wn * 32 + l16;
    int nB = nA + 16;
    float vA = acc0.f[j] + be[(size_t)e * I_ + nA];
    float vB = acc1.f[j] + be[(size_t)e * I_ + nB];
    vA = 0.5f * vA * (1.0f + erff(vA * 0.70710678f));
    vB = 0.5f * vB * (1.0f + erff(vB * 0.70710678f));
    inter[(size_t)tok * I_ + nA] = f2bf(vA);
    inter[(size_t)tok * I_ + nB] = f2bf(vB);
  }
}

// ---------------------------------------------------------------------------
extern "C" void kernel_launch(void* const* d_in, const int* in_sizes, int n_in,
                              void* d_out, int out_size, void* d_ws, size_t ws_size,
                              hipStream_t stream)
{
  const float* x     = (const float*)d_in[0];
  const float* Wq    = (const float*)d_in[1];
  const float* bq    = (const float*)d_in[2];
  const float* Wk    = (const float*)d_in[3];
  const float* bk    = (const float*)d_in[4];
  const float* Wv    = (const float*)d_in[5];
  const float* bv    = (const float*)d_in[6];
  const float* Wao   = (const float*)d_in[7];
  const float* bao   = (const float*)d_in[8];
  const float* ln1g  = (const float*)d_in[9];
  const float* ln1b  = (const float*)d_in[10];
  const float* rw    = (const float*)d_in[11];
  const float* We    = (const float*)d_in[12];
  const float* be    = (const float*)d_in[13];
  const float* Wo    = (const float*)d_in[14];
  const float* bo    = (const float*)d_in[15];
  const float* ln2g  = (const float*)d_in[16];
  const float* ln2b  = (const float*)d_in[17];
  (void)in_sizes; (void)n_in; (void)out_size; (void)ws_size;

  char* ws = (char*)d_ws;
  size_t off = 0;
  auto take = [&](size_t bytes) {
    char* p = ws + off; off += (bytes + 255) & ~(size_t)255; return p;
  };
  __bf16* qb   = (__bf16*)take((size_t)N_ * H_ * 2);
  __bf16* kb   = (__bf16*)take((size_t)N_ * H_ * 2);
  __bf16* vb   = (__bf16*)take((size_t)N_ * H_ * 2);
  float*  ctx  = (float*)take((size_t)N_ * H_ * 4);
  // inter (32MB bf16) aliases q/k/v/ctx (40MB), which are dead before experts run
  __bf16* inter = (__bf16*)ws;
  float*  tmp  = (float*)take((size_t)N_ * H_ * 4);
  float*  aout = (float*)take((size_t)N_ * H_ * 4);
  int* idx     = (int*)take((size_t)N_ * 4);
  int* counts  = (int*)take(64);
  int* offs    = (int*)take(64);
  int* cursor  = (int*)take(64);
  int* perm    = (int*)take((size_t)N_ * 4);

  const float qscale = 0.125f * 1.44269504088896f;   // 1/sqrt(DH) * log2(e)

  dim3 blk(256);
  dim3 gQKV(H_ / 64, N_ / 64);
  k_gemm<false, true, false><<<gQKV, blk, 0, stream>>>(x, Wq, bq, nullptr, qb, N_, H_, H_, qscale);
  k_gemm<false, true, false><<<gQKV, blk, 0, stream>>>(x, Wk, bk, nullptr, kb, N_, H_, H_, 1.0f);
  k_gemm<false, true, false><<<gQKV, blk, 0, stream>>>(x, Wv, bv, nullptr, vb, N_, H_, H_, 1.0f);

  k_attn<<<dim3(S_ / 128, NH_, B_), blk, 0, stream>>>(qb, kb, vb, ctx);

  k_gemm<false, false, true><<<gQKV, blk, 0, stream>>>(ctx, Wao, bao, x, tmp, N_, H_, H_, 1.0f);
  k_ln<<<dim3(N_), blk, 0, stream>>>(tmp, ln1g, ln1b, aout);

  k_zero16<<<1, 32, 0, stream>>>(counts);
  k_router<<<dim3(N_ / 256), blk, 0, stream>>>(aout, rw, idx, counts);
  k_offsets<<<1, 32, 0, stream>>>(counts, offs, cursor);
  k_scatter<<<dim3(N_ / 256), blk, 0, stream>>>(idx, cursor, perm);

  k_expert<<<dim3(I_ / 64, N_ / 64, E_), blk, 0, stream>>>(aout, We, be, perm, offs,
                                                           counts, inter);

  k_gemm<true, false, true><<<dim3(H_ / 64, N_ / 64), blk, 0, stream>>>(
      inter, Wo, bo, aout, tmp, N_, H_, I_, 1.0f);
  k_ln<<<dim3(N_), blk, 0, stream>>>(tmp, ln2g, ln2b, (float*)d_out);
}
